// MultiwayTransformerEncoder_79998060855669
// MI455X (gfx1250) — matrix operand order repaired and checked by reading, not compiled, not run on hardware
//
#include <hip/hip_runtime.h>

// ---------------- feature detection: CDNA5 async global->LDS ----------------
#if defined(__has_builtin)
#  if __has_builtin(__builtin_amdgcn_global_load_async_to_lds_b128) && \
      __has_builtin(__builtin_amdgcn_s_wait_asynccnt)
#    define USE_ASYNC_LDS 1
#  endif
#endif
#ifndef USE_ASYNC_LDS
#  define USE_ASYNC_LDS 0
#endif

// ---------------- problem constants ----------------
constexpr int kL = 2, kC = 1024, kH = 16, kI = 4096, kE = 8, kB = 4, kN = 640;
constexpr int kHD = kC / kH;   // 64
constexpr int kR  = kB * kN;   // 2560 token rows
constexpr float kScale = 0.125f;   // HD^-0.5
constexpr float kEps = 1e-6f;

// ---------------- types for WMMA ----------------
typedef __attribute__((ext_vector_type(16))) __bf16          v16bf;
typedef __attribute__((ext_vector_type(8)))  float           v8f;
typedef __attribute__((ext_vector_type(8)))  unsigned short  v8u;

union FragBF {
  v16bf bf;
  v8u   h[2];
};

__device__ __forceinline__ unsigned short f2bf(float f) {
  unsigned int u = __float_as_uint(f);
  u += 0x7FFFu + ((u >> 16) & 1u);          // round-to-nearest-even
  return (unsigned short)(u >> 16);
}
__device__ __forceinline__ float bf2f(unsigned short u) {
  return __uint_as_float(((unsigned int)u) << 16);
}

#if USE_ASYNC_LDS
// builtin signature (from compiler diagnostic): takes pointers to 128-bit int
// vectors — (v4i addrspace(1)*, v4i addrspace(3)*, imm offset, imm cpol)
typedef int v4i_ __attribute__((vector_size(16)));
typedef __attribute__((address_space(1))) v4i_ as1_v4i;
typedef __attribute__((address_space(3))) v4i_ as3_v4i;

__device__ __forceinline__ void async_ld_b128(const void* g, void* l) {
  __builtin_amdgcn_global_load_async_to_lds_b128(
      (as1_v4i*)g, (as3_v4i*)l, 0, 0);
}
#endif

// =====================================================================
// Tiled WMMA bf16 GEMM:  C[M,N](f32) = A[M,K](bf16) * B[N,K]^T(bf16) + bias[N]
// Block tile 128 x TNP, K-step 32, 256 threads = 8 waves (4x2),
// wave tile 32 x (TNP/2) => 2 x (TNP/32) fragments of v_wmma_f32_16x16x32_bf16.
// Async path: 2-deep LDS ping-pong fed by global_load_async_to_lds_b128.
// Fallback: register-staged double buffer.
// Requires M % 128 == 0, N % TNP == 0, K % 32 == 0 (true for all uses here).
// =====================================================================
constexpr int TM = 128, TK = 32;
constexpr int LDSS = TK + 8;   // 40 ushorts = 80B row: 16B aligned, bank-staggered

template <int TNP>
__global__ __launch_bounds__(256)
void k_gemm_bf16(const unsigned short* __restrict__ Abase,
                 const unsigned short* __restrict__ Bbase,
                 float* __restrict__ Cbase,
                 const float* __restrict__ bias,
                 int M, int N, int K,
                 long long strideA, long long strideB, long long strideC)
{
  constexpr int NF  = TNP / 32;   // B fragments per wave
  constexpr int BCH = TNP / 64;   // B chunks (8 bf16) per thread per tile

  const unsigned short* A = Abase + (long long)blockIdx.z * strideA;
  const unsigned short* B = Bbase + (long long)blockIdx.z * strideB;
  float*                C = Cbase + (long long)blockIdx.z * strideC;

  const int tid  = threadIdx.x;
  const int lane = tid & 31;          // wave32
  const int wid  = tid >> 5;          // 8 waves
  const int wM   = wid >> 1;          // 0..3
  const int wN   = wid & 1;           // 0..1
  const int m0   = blockIdx.y * TM;
  const int n0   = blockIdx.x * TNP;

  const v8f vzero = {0.f,0.f,0.f,0.f,0.f,0.f,0.f,0.f};
  v8f acc[2][NF];
#pragma unroll
  for (int i = 0; i < 2; ++i)
#pragma unroll
    for (int j = 0; j < NF; ++j) acc[i][j] = vzero;

  const int lr  = lane & 15;
  const int kbA = (lane >> 4) * 8;    // A frag: lanes 0-15 K base 0, 16-31 base 8
  const int kbB = (lane >> 4) * 8;    // B frag: lanes 0-15 K base 0, 16-31 base 8

  auto compute = [&](const unsigned short* as, const unsigned short* bs) {
    FragBF fa[2];
    FragBF fb[NF];
#pragma unroll
    for (int i = 0; i < 2; ++i) {            // A 16x32 bf16 fragment layout
      int row = wM * 32 + i * 16 + lr;
      fa[i].h[0] = *(const v8u*)&as[row * LDSS + kbA];
      fa[i].h[1] = *(const v8u*)&as[row * LDSS + kbA + 16];
    }
#pragma unroll
    for (int j = 0; j < NF; ++j) {           // B 32x16 bf16 fragment layout
      int row = wN * (TNP / 2) + j * 16 + lr;
      fb[j].h[0] = *(const v8u*)&bs[row * LDSS + kbB];
      fb[j].h[1] = *(const v8u*)&bs[row * LDSS + kbB + 16];
    }
#pragma unroll
    for (int i = 0; i < 2; ++i)
#pragma unroll
      for (int j = 0; j < NF; ++j)
        acc[i][j] = __builtin_amdgcn_wmma_f32_16x16x32_bf16(
            false, fa[i].bf, false, fb[j].bf, (short)0, acc[i][j], false, false);
  };

#if USE_ASYNC_LDS
  // ---- async global->LDS DMA with 2-deep ping-pong ----
  __shared__ unsigned short As[2][TM * LDSS];
  __shared__ unsigned short Bs[2][TNP * LDSS];

  auto issue = [&](int k0, int b) {
#pragma unroll
    for (int t = 0; t < 2; ++t) {            // A tile: 512 chunks / 256 thr
      int q = tid + t * 256;
      int r = q >> 2, c8 = (q & 3) << 3;
      async_ld_b128(&A[(long long)(m0 + r) * K + k0 + c8],
                    &As[b][r * LDSS + c8]);
    }
#pragma unroll
    for (int t = 0; t < BCH; ++t) {          // B tile
      int q = tid + t * 256;
      int r = q >> 2, c8 = (q & 3) << 3;
      async_ld_b128(&B[(long long)(n0 + r) * K + k0 + c8],
                    &Bs[b][r * LDSS + c8]);
    }
  };

  issue(0, 0);
  int buf = 0;
  for (int k0 = 0; k0 < K; k0 += TK) {
    __builtin_amdgcn_s_wait_asynccnt(0);     // my DMA for tile k0 done
    __syncthreads();                         // everyone's DMA done + prior reads retired
    if (k0 + TK < K) issue(k0 + TK, buf ^ 1);  // overlap next DMA with WMMAs
    compute(As[buf], Bs[buf]);
    buf ^= 1;
  }
#else
  // ---- fallback: register-staged double buffer ----
  __shared__ unsigned short As[TM * LDSS];
  __shared__ unsigned short Bs[TNP * LDSS];

  v8u ra[2];
  v8u rb[BCH];
  auto loadg = [&](int k0) {
#pragma unroll
    for (int t = 0; t < 2; ++t) {
      int q = tid + t * 256;
      int r = q >> 2, c8 = (q & 3) << 3;
      ra[t] = *(const v8u*)&A[(long long)(m0 + r) * K + k0 + c8];
    }
#pragma unroll
    for (int t = 0; t < BCH; ++t) {
      int q = tid + t * 256;
      int r = q >> 2, c8 = (q & 3) << 3;
      rb[t] = *(const v8u*)&B[(long long)(n0 + r) * K + k0 + c8];
    }
    __builtin_prefetch(&A[(long long)(m0 + (tid & (TM - 1))) * K + k0], 0, 0);
    __builtin_prefetch(&B[(long long)(n0 + (tid & (TNP - 1))) * K + k0], 0, 0);
  };

  loadg(0);
  for (int k0 = 0; k0 < K; k0 += TK) {
    __syncthreads();
#pragma unroll
    for (int t = 0; t < 2; ++t) {
      int q = tid + t * 256;
      int r = q >> 2, c8 = (q & 3) << 3;
      *(v8u*)&As[r * LDSS + c8] = ra[t];
    }
#pragma unroll
    for (int t = 0; t < BCH; ++t) {
      int q = tid + t * 256;
      int r = q >> 2, c8 = (q & 3) << 3;
      *(v8u*)&Bs[r * LDSS + c8] = rb[t];
    }
    __syncthreads();
    if (k0 + TK < K) loadg(k0 + TK);         // overlap next loads with WMMAs
    compute(As, Bs);
  }
#endif

  // epilogue: C/D layout — lane<16: M=r, N=lane; lane>=16: M=r+8, N=lane-16
  const int rofs = (lane >> 4) * 8;
#pragma unroll
  for (int i = 0; i < 2; ++i) {
#pragma unroll
    for (int j = 0; j < NF; ++j) {
      int mb  = m0 + wM * 32 + i * 16 + rofs;
      int col = n0 + wN * (TNP / 2) + j * 16 + lr;
      float bv = bias ? bias[col] : 0.0f;
#pragma unroll
      for (int r = 0; r < 8; ++r)
        C[(long long)(mb + r) * N + col] = acc[i][j][r] + bv;
    }
  }
}

// =====================================================================
// Elementwise / reduction helper kernels
// =====================================================================
__global__ void k_f32_to_bf16(const float* __restrict__ in,
                              unsigned short* __restrict__ out, long long n) {
  for (long long i = (long long)blockIdx.x * blockDim.x + threadIdx.x; i < n;
       i += (long long)gridDim.x * blockDim.x)
    out[i] = f2bf(in[i]);
}

__global__ void k_copy_f32(const float* __restrict__ in, float* __restrict__ out,
                           long long n) {
  for (long long i = (long long)blockIdx.x * blockDim.x + threadIdx.x; i < n;
       i += (long long)gridDim.x * blockDim.x)
    out[i] = in[i];
}

__global__ void k_zero_f32(float* __restrict__ p, long long n) {
  for (long long i = (long long)blockIdx.x * blockDim.x + threadIdx.x; i < n;
       i += (long long)gridDim.x * blockDim.x)
    p[i] = 0.0f;
}

// x += h * gamma[col]
__global__ void k_residual(float* __restrict__ x, const float* __restrict__ h,
                           const float* __restrict__ g, long long n) {
  for (long long i = (long long)blockIdx.x * blockDim.x + threadIdx.x; i < n;
       i += (long long)gridDim.x * blockDim.x)
    x[i] += h[i] * g[i & (kC - 1)];
}

// exact GELU then bf16
__global__ void k_gelu_bf16(const float* __restrict__ h,
                            unsigned short* __restrict__ out, long long n) {
  for (long long i = (long long)blockIdx.x * blockDim.x + threadIdx.x; i < n;
       i += (long long)gridDim.x * blockDim.x) {
    float x = h[i];
    out[i] = f2bf(0.5f * x * (1.0f + erff(x * 0.70710678118654752f)));
  }
}

// acc += combine[row][e] * y
__global__ void k_combine_acc(float* __restrict__ acc, const float* __restrict__ y,
                              const float* __restrict__ comb, int e, long long n) {
  for (long long i = (long long)blockIdx.x * blockDim.x + threadIdx.x; i < n;
       i += (long long)gridDim.x * blockDim.x)
    acc[i] += comb[(i >> 10) * kE + e] * y[i];
}

// LayerNorm over C=1024, one block per token row, bf16 output
__global__ __launch_bounds__(256)
void k_layernorm_bf16(const float* __restrict__ x, const float* __restrict__ s,
                      const float* __restrict__ b, unsigned short* __restrict__ out) {
  __shared__ float r1[256];
  __shared__ float r2[256];
  const long long row = blockIdx.x;
  const float* xr = x + row * kC;
  float v[4], sum = 0.f, sq = 0.f;
#pragma unroll
  for (int i = 0; i < 4; ++i) {
    v[i] = xr[threadIdx.x + i * 256];
    sum += v[i];
    sq  += v[i] * v[i];
  }
  r1[threadIdx.x] = sum; r2[threadIdx.x] = sq;
  __syncthreads();
  for (int off = 128; off > 0; off >>= 1) {
    if (threadIdx.x < off) {
      r1[threadIdx.x] += r1[threadIdx.x + off];
      r2[threadIdx.x] += r2[threadIdx.x + off];
    }
    __syncthreads();
  }
  float mean = r1[0] / (float)kC;
  float var  = r2[0] / (float)kC - mean * mean;
  float rstd = rsqrtf(var + kEps);
#pragma unroll
  for (int i = 0; i < 4; ++i) {
    int c = threadIdx.x + i * 256;
    out[row * kC + c] = f2bf((v[i] - mean) * rstd * s[c] + b[c]);
  }
}

// split qkv [R,3C] -> q(scaled,+qb)[B,H,N,HD], k[B,H,N,HD], v^T(+vb)[B,H,HD,N], all bf16
__global__ void k_qkv_split(const float* __restrict__ qkv,
                            const float* __restrict__ qb, const float* __restrict__ vb,
                            unsigned short* __restrict__ q, unsigned short* __restrict__ k,
                            unsigned short* __restrict__ vT) {
  const long long total = (long long)kB * kH * kN * kHD;
  long long i = (long long)blockIdx.x * blockDim.x + threadIdx.x;
  if (i >= total) return;
  int d = (int)(i & (kHD - 1));
  long long t = i >> 6;                 // /kHD
  int n = (int)(t % kN); t /= kN;
  int h = (int)(t % kH);
  int b = (int)(t / kH);
  long long row = (long long)(b * kN + n) * (3 * kC);
  int cq = h * kHD + d;
  q[i] = f2bf((qkv[row + cq] + qb[cq]) * kScale);
  k[i] = f2bf(qkv[row + kC + cq]);
  vT[((long long)(b * kH + h) * kHD + d) * kN + n] = f2bf(qkv[row + 2 * kC + cq] + vb[cq]);
}

// scores row -> +rpb, mask, softmax, bf16 P. One block per (b,h,q) row.
__global__ __launch_bounds__(256)
void k_softmax_bf16(float* __restrict__ S, const float* __restrict__ rpb,
                    const int* __restrict__ mask, unsigned short* __restrict__ P) {
  __shared__ float red[256];
  const int r = blockIdx.x;                 // (b*H + h)*N + q
  const int q = r % kN;
  const int h = (r / kN) % kH;
  const int b = r / (kN * kH);
  float* Sr = S + (long long)r * kN;
  const float* rr = rpb + ((long long)h * kN + q) * kN;

  float mx = -1e30f;
  for (int k = threadIdx.x; k < kN; k += 256) {
    float v = Sr[k] + rr[k];
    if (mask[b * kN + k] == 0) v = -1e30f;
    Sr[k] = v;
    mx = fmaxf(mx, v);
  }
  red[threadIdx.x] = mx;
  __syncthreads();
  for (int off = 128; off > 0; off >>= 1) {
    if (threadIdx.x < off)
      red[threadIdx.x] = fmaxf(red[threadIdx.x], red[threadIdx.x + off]);
    __syncthreads();
  }
  mx = red[0];
  __syncthreads();

  float sum = 0.f;
  for (int k = threadIdx.x; k < kN; k += 256) {
    float e = expf(Sr[k] - mx);
    Sr[k] = e;
    sum += e;
  }
  red[threadIdx.x] = sum;
  __syncthreads();
  for (int off = 128; off > 0; off >>= 1) {
    if (threadIdx.x < off) red[threadIdx.x] += red[threadIdx.x + off];
    __syncthreads();
  }
  float inv = 1.0f / red[0];
  for (int k = threadIdx.x; k < kN; k += 256)
    P[(long long)r * kN + k] = f2bf(Sr[k] * inv);
}

// merge heads: attn [B,H,N,HD] f32 -> [R,C] bf16
__global__ void k_merge_heads(const float* __restrict__ attn,
                              unsigned short* __restrict__ out, long long n) {
  for (long long i = (long long)blockIdx.x * blockDim.x + threadIdx.x; i < n;
       i += (long long)gridDim.x * blockDim.x) {
    int c = (int)(i & (kC - 1));
    long long row = i >> 10;
    int b  = (int)(row / kN);
    int nn = (int)(row % kN);
    int h = c / kHD, d = c & (kHD - 1);
    out[i] = f2bf(attn[((long long)(b * kH + h) * kN + nn) * kHD + d]);
  }
}

// top-2 softmax gating; one block per token row (8 experts x 32 lanes)
__global__ __launch_bounds__(256)
void k_gate(const unsigned short* __restrict__ xbf, const float* __restrict__ gw,
            float* __restrict__ comb, float* __restrict__ me_acc,
            float* __restrict__ ce_acc) {
  __shared__ float sh[256];
  __shared__ float lg[kE];
  const int row = blockIdx.x;
  const int e = threadIdx.x >> 5;
  const int l = threadIdx.x & 31;
  const unsigned short* xr = xbf + (long long)row * kC;
  const float* we = gw + e * kC;
  float p = 0.f;
  for (int c = l; c < kC; c += 32) p += bf2f(xr[c]) * we[c];
  sh[threadIdx.x] = p;
  __syncthreads();
  if (threadIdx.x < kE) {
    float s = 0.f;
    for (int i = 0; i < 32; ++i) s += sh[threadIdx.x * 32 + i];
    lg[threadIdx.x] = s;
  }
  __syncthreads();
  if (threadIdx.x == 0) {
    float probs[kE];
    float mx = lg[0];
    for (int i = 1; i < kE; ++i) mx = fmaxf(mx, lg[i]);
    float sum = 0.f;
    for (int i = 0; i < kE; ++i) { probs[i] = expf(lg[i] - mx); sum += probs[i]; }
    for (int i = 0; i < kE; ++i) probs[i] /= sum;
    int i0 = 0;
    for (int i = 1; i < kE; ++i) if (probs[i] > probs[i0]) i0 = i;
    int i1 = (i0 == 0) ? 1 : 0;
    for (int i = 0; i < kE; ++i) if (i != i0 && probs[i] > probs[i1]) i1 = i;
    float g0 = probs[i0], g1 = probs[i1];
    float inv = 1.0f / (g0 + g1);
    for (int i = 0; i < kE; ++i) comb[(long long)row * kE + i] = 0.f;
    comb[(long long)row * kE + i0] = g0 * inv;
    comb[(long long)row * kE + i1] = g1 * inv;
    for (int i = 0; i < kE; ++i) atomicAdd(&me_acc[i], probs[i]);
    atomicAdd(&ce_acc[i0], 1.0f);
    atomicAdd(&ce_acc[i1], 1.0f);
  }
}

// aux = sum_l E * sum_e (me/R)*(ce/R)
__global__ void k_aux(const float* __restrict__ me, const float* __restrict__ ce,
                      float* __restrict__ out) {
  if (blockIdx.x == 0 && threadIdx.x == 0) {
    float aux = 0.f;
    for (int l = 0; l < kL; ++l) {
      float s = 0.f;
      for (int e = 0; e < kE; ++e)
        s += (me[l * kE + e] / (float)kR) * (ce[l * kE + e] / (float)kR);
      aux += (float)kE * s;
    }
    out[0] = aux;
  }
}

// =====================================================================
static inline int gs_blocks(long long n) {
  long long b = (n + 255) / 256;
  return (int)(b > 16384 ? 16384 : (b < 1 ? 1 : b));
}

extern "C" void kernel_launch(void* const* d_in, const int* in_sizes, int n_in,
                              void* d_out, int out_size, void* d_ws, size_t ws_size,
                              hipStream_t stream) {
  (void)in_sizes; (void)n_in; (void)out_size; (void)ws_size;
  const float* hidden = (const float*)d_in[0];
  const int*   mask   = (const int*)d_in[1];
  const float* rpb    = (const float*)d_in[2];
  const float* qkv_w  = (const float*)d_in[3];
  const float* q_bias = (const float*)d_in[4];
  const float* v_bias = (const float*)d_in[5];
  const float* proj_w = (const float*)d_in[6];
  const float* proj_b = (const float*)d_in[7];
  const float* ln1_s  = (const float*)d_in[8];
  const float* ln1_b  = (const float*)d_in[9];
  const float* ln2_s  = (const float*)d_in[10];
  const float* ln2_b  = (const float*)d_in[11];
  const float* gamma1 = (const float*)d_in[12];
  const float* gamma2 = (const float*)d_in[13];
  const float* gate_w = (const float*)d_in[14];
  const float* fc1_w  = (const float*)d_in[15];
  const float* fc1_b  = (const float*)d_in[16];
  const float* fc2_w  = (const float*)d_in[17];
  const float* fc2_b  = (const float*)d_in[18];

  // ---- workspace carve-out ----
  size_t off = 0;
  auto alloc = [&](size_t bytes) -> void* {
    void* p = (char*)d_ws + off;
    off = (off + bytes + 255) & ~(size_t)255;
    return p;
  };
  const long long nQW = (long long)kL * 3 * kC * kC;
  const long long nPW = (long long)kL * kC * kC;
  const long long nF1 = (long long)kL * kE * kI * kC;
  const long long nF2 = (long long)kL * kE * kC * kI;
  const long long nBH = (long long)kB * kH * kN * kHD;   // per-head act elems
  const long long nSC = (long long)kB * kH * kN * kN;    // score elems

  unsigned short* wq  = (unsigned short*)alloc((size_t)nQW * 2);
  unsigned short* wp  = (unsigned short*)alloc((size_t)nPW * 2);
  unsigned short* wf1 = (unsigned short*)alloc((size_t)nF1 * 2);
  unsigned short* wf2 = (unsigned short*)alloc((size_t)nF2 * 2);
  float*          xcur = (float*)alloc((size_t)kR * kC * 4);
  unsigned short* xbf  = (unsigned short*)alloc((size_t)kR * kC * 2);
  float*          qkvb = (float*)alloc((size_t)kR * 3 * kC * 4);
  unsigned short* qh   = (unsigned short*)alloc((size_t)nBH * 2);
  unsigned short* kh   = (unsigned short*)alloc((size_t)nBH * 2);
  unsigned short* vT   = (unsigned short*)alloc((size_t)nBH * 2);
  float*          scores = (float*)alloc((size_t)nSC * 4);
  unsigned short* Pm     = (unsigned short*)alloc((size_t)nSC * 2);
  float*          attno  = (float*)alloc((size_t)nBH * 4);
  unsigned short* abf    = (unsigned short*)alloc((size_t)kR * kC * 2);
  float*          hbuf   = (float*)alloc((size_t)kR * kI * 4);
  unsigned short* hbf    = (unsigned short*)alloc((size_t)kR * kI * 2);
  float*          ybuf   = (float*)alloc((size_t)kR * kC * 4);
  float*          macc   = (float*)alloc((size_t)kR * kC * 4);
  float*          comb   = (float*)alloc((size_t)kR * kE * 4);
  float*          mece   = (float*)alloc((size_t)2 * kL * kE * 4);
  float* me = mece;
  float* ce = mece + kL * kE;

  // ---- weights -> bf16 (deterministic, every launch) ----
  k_f32_to_bf16<<<gs_blocks(nQW), 256, 0, stream>>>(qkv_w,  wq,  nQW);
  k_f32_to_bf16<<<gs_blocks(nPW), 256, 0, stream>>>(proj_w, wp,  nPW);
  k_f32_to_bf16<<<gs_blocks(nF1), 256, 0, stream>>>(fc1_w,  wf1, nF1);
  k_f32_to_bf16<<<gs_blocks(nF2), 256, 0, stream>>>(fc2_w,  wf2, nF2);

  k_copy_f32<<<gs_blocks((long long)kR * kC), 256, 0, stream>>>(hidden, xcur,
                                                                (long long)kR * kC);
  k_zero_f32<<<1, 64, 0, stream>>>(mece, (long long)2 * kL * kE);

  const long long nRC = (long long)kR * kC;
  for (int l = 0; l < kL; ++l) {
    const unsigned short* wql = wq + (size_t)l * 3 * kC * kC;
    const unsigned short* wpl = wp + (size_t)l * kC * kC;

    // --- attention half ---
    k_layernorm_bf16<<<kR, 256, 0, stream>>>(xcur, ln1_s + l * kC, ln1_b + l * kC, xbf);

    dim3 gqkv(3 * kC / 128, kR / TM, 1);
    k_gemm_bf16<128><<<gqkv, 256, 0, stream>>>(xbf, wql, qkvb, nullptr,
                                               kR, 3 * kC, kC, 0, 0, 0);

    k_qkv_split<<<(int)((nBH + 255) / 256), 256, 0, stream>>>(
        qkvb, q_bias + l * kC, v_bias + l * kC, qh, kh, vT);

    dim3 gsc(kN / 128, kN / TM, kB * kH);   // S = Q K^T  (batched over b,h)
    k_gemm_bf16<128><<<gsc, 256, 0, stream>>>(qh, kh, scores, nullptr,
                                              kN, kN, kHD,
                                              (long long)kN * kHD, (long long)kN * kHD,
                                              (long long)kN * kN);

    k_softmax_bf16<<<kB * kH * kN, 256, 0, stream>>>(scores, rpb, mask, Pm);

    dim3 gav(kHD / 64, kN / TM, kB * kH);   // O = P V  (B = V^T stored [HD,N])
    k_gemm_bf16<64><<<gav, 256, 0, stream>>>(Pm, vT, attno, nullptr,
                                             kN, kHD, kN,
                                             (long long)kN * kN, (long long)kHD * kN,
                                             (long long)kN * kHD);

    k_merge_heads<<<gs_blocks(nRC), 256, 0, stream>>>(attno, abf, nRC);

    dim3 gpr(kC / 128, kR / TM, 1);
    k_gemm_bf16<128><<<gpr, 256, 0, stream>>>(abf, wpl, ybuf, proj_b + l * kC,
                                              kR, kC, kC, 0, 0, 0);

    k_residual<<<gs_blocks(nRC), 256, 0, stream>>>(xcur, ybuf, gamma1 + l * kC, nRC);

    // --- MoE half ---
    k_layernorm_bf16<<<kR, 256, 0, stream>>>(xcur, ln2_s + l * kC, ln2_b + l * kC, xbf);

    k_gate<<<kR, 256, 0, stream>>>(xbf, gate_w + (size_t)l * kE * kC, comb,
                                   me + l * kE, ce + l * kE);

    k_zero_f32<<<gs_blocks(nRC), 256, 0, stream>>>(macc, nRC);

    for (int e = 0; e < kE; ++e) {
      const unsigned short* w1 = wf1 + (size_t)(l * kE + e) * kI * kC;
      const unsigned short* w2 = wf2 + (size_t)(l * kE + e) * kC * kI;

      dim3 gf1(kI / 128, kR / TM, 1);
      k_gemm_bf16<128><<<gf1, 256, 0, stream>>>(xbf, w1, hbuf,
                                                fc1_b + (size_t)(l * kE + e) * kI,
                                                kR, kI, kC, 0, 0, 0);

      k_gelu_bf16<<<gs_blocks((long long)kR * kI), 256, 0, stream>>>(
          hbuf, hbf, (long long)kR * kI);

      dim3 gf2(kC / 128, kR / TM, 1);
      k_gemm_bf16<128><<<gf2, 256, 0, stream>>>(hbf, w2, ybuf,
                                                fc2_b + (size_t)(l * kE + e) * kC,
                                                kR, kC, kI, 0, 0, 0);

      k_combine_acc<<<gs_blocks(nRC), 256, 0, stream>>>(macc, ybuf, comb, e, nRC);
    }

    k_residual<<<gs_blocks(nRC), 256, 0, stream>>>(xcur, macc, gamma2 + l * kC, nRC);
  }

  // ---- outputs: x then aux scalar ----
  k_copy_f32<<<gs_blocks(nRC), 256, 0, stream>>>(xcur, (float*)d_out, nRC);
  k_aux<<<1, 32, 0, stream>>>(me, ce, (float*)d_out + nRC);
}
